// SelfAttention_87651692577365
// MI455X (gfx1250) — matrix operand order repaired and checked by reading, not compiled, not run on hardware
//
#include <hip/hip_runtime.h>

#define B_N 8
#define S_N 2048
#define D_N 512
#define ROWS_N (B_N * S_N)   // 16384

typedef __attribute__((ext_vector_type(16))) __bf16 v16bf;
typedef __attribute__((ext_vector_type(8)))  float  v8f;

union FragBF {
    v16bf v;
    unsigned short us[16];
    uint4 q[2];
};

union Pack8 {
    uint4 q;
    unsigned short us[8];
};

// fp32 -> bf16, round-to-nearest-even
__device__ __forceinline__ unsigned short f2bf(float f) {
    union { float f; unsigned int u; } c;
    c.f = f;
    unsigned int u = c.u;
    u += 0x7FFFu + ((u >> 16) & 1u);
    return (unsigned short)(u >> 16);
}

// ---------------------------------------------------------------------------
// Kernel 1: Y = X * W + b   (X: [16384,512] f32, W: [512,512] f32, Y bf16)
// Block tile 128(M) x 64(N), 256 threads = 8 waves, each wave: 16 rows x 64 cols.
// TR=false: row-major out [row][col] (Q, K).
// TR=true : feature-major out [b][col][s] (V^T) -- one b128 store per lane/tile,
//           so the attention kernel can read V fragments key-contiguously.
// ---------------------------------------------------------------------------
template <bool TR>
__global__ __launch_bounds__(256) void qkv_proj_kernel(
    const float* __restrict__ X, const float* __restrict__ W,
    const float* __restrict__ bias, unsigned short* __restrict__ Out)
{
    __shared__ __align__(16) unsigned short Abuf[128 * 32]; // [row][k] bf16
    __shared__ __align__(16) unsigned short Bt[64 * 32];    // [col][k] bf16 (W transposed)

    const int tid  = threadIdx.x;
    const int wave = tid >> 5;
    const int lane = tid & 31;
    const int lo   = lane & 15;
    const int hi   = lane >> 4;
    const int rowBase = blockIdx.y * 128;
    const int colBase = blockIdx.x * 64;

    const v8f zf = {0.f,0.f,0.f,0.f,0.f,0.f,0.f,0.f};
    v8f acc[4] = {zf, zf, zf, zf};

    for (int kk = 0; kk < D_N / 32; ++kk) {
        __syncthreads();
        // Stage A tile: 128x32 f32 -> bf16 (1024 float4, 4 per thread)
        #pragma unroll
        for (int j = 0; j < 4; ++j) {
            int idx = tid * 4 + j;            // 0..1023
            int r   = idx >> 3;               // row 0..127
            int c4  = idx & 7;                // float4 group
            float4 f = *(const float4*)(X + (size_t)(rowBase + r) * D_N + kk * 32 + c4 * 4);
            unsigned short* dst = &Abuf[r * 32 + c4 * 4];
            dst[0] = f2bf(f.x); dst[1] = f2bf(f.y); dst[2] = f2bf(f.z); dst[3] = f2bf(f.w);
        }
        // Stage W tile transposed: rows kk*32..+32, cols colBase..+64 (512 float4, 2/thread)
        #pragma unroll
        for (int j = 0; j < 2; ++j) {
            int idx = tid * 2 + j;            // 0..511
            int r   = idx >> 4;               // k 0..31
            int c4  = idx & 15;               // col group of 4
            float4 f = *(const float4*)(W + (size_t)(kk * 32 + r) * D_N + colBase + c4 * 4);
            int c = c4 * 4;
            Bt[(c + 0) * 32 + r] = f2bf(f.x);
            Bt[(c + 1) * 32 + r] = f2bf(f.y);
            Bt[(c + 2) * 32 + r] = f2bf(f.z);
            Bt[(c + 3) * 32 + r] = f2bf(f.w);
        }
        __syncthreads();

        // A fragment: row = wave*16 + lo, contiguous K run (hi*16 .. hi*16+15)
        FragBF a;
        const unsigned short* ap = &Abuf[(wave * 16 + lo) * 32];
        a.q[0] = *(const uint4*)(ap + hi * 16);
        a.q[1] = *(const uint4*)(ap + hi * 16 + 8);

        #pragma unroll
        for (int nt = 0; nt < 4; ++nt) {
            FragBF b;
            const unsigned short* bp = &Bt[(nt * 16 + lo) * 32 + hi * 16];
            b.q[0] = *(const uint4*)(bp);
            b.q[1] = *(const uint4*)(bp + 8);
            acc[nt] = __builtin_amdgcn_wmma_f32_16x16x32_bf16(
                false, a.v, false, b.v, (short)0, acc[nt], false, false);
        }
    }

    // Epilogue: + bias, convert to bf16, store. C layout: M = r + 8*hi, N = lo.
    const int row0 = rowBase + wave * 16 + hi * 8;   // 8 consecutive rows r=0..7
    if (TR) {
        const int b  = row0 >> 11;        // /2048 (tiles never cross batch)
        const int s0 = row0 & 2047;
        #pragma unroll
        for (int nt = 0; nt < 4; ++nt) {
            int c = colBase + nt * 16 + lo;
            float bb = bias[c];
            Pack8 p;
            #pragma unroll
            for (int r = 0; r < 8; ++r) p.us[r] = f2bf(acc[nt][r] + bb);
            *(uint4*)(Out + ((size_t)b * D_N + c) * S_N + s0) = p.q;
        }
    } else {
        #pragma unroll
        for (int nt = 0; nt < 4; ++nt) {
            int c = colBase + nt * 16 + lo;
            float bb = bias[c];
            #pragma unroll
            for (int r = 0; r < 8; ++r) {
                float v = acc[nt][r] + bb;
                Out[(size_t)(row0 + r) * D_N + c] = f2bf(v);
            }
        }
    }
}

// ---------------------------------------------------------------------------
// Kernel 2: flash attention. One 16-query tile per block (256 thr = 8 waves).
// Each wave owns a 64-wide slice of the 512-dim output (16x64 f32 accum).
// Q tile staged via GLOBAL_LOAD_ASYNC_TO_LDS_B128 (ASYNCcnt); K and V^T
// fragments read straight from global (hot in 192MB L2: 4MB per batch).
// ---------------------------------------------------------------------------
__global__ __launch_bounds__(256) void attn_kernel(
    const unsigned short* __restrict__ Qws,   // [b*S][D] bf16
    const unsigned short* __restrict__ Kws,   // [b*S][D] bf16
    const unsigned short* __restrict__ Vtws,  // [b][D][S] bf16 (V transposed)
    const float* __restrict__ maskp,
    float* __restrict__ Out)
{
    __shared__ __align__(16) unsigned short Qtile[16 * D_N]; // 16KB bf16
    __shared__ __align__(16) float          Sfull[16 * 32];  // 2KB scores
    __shared__ __align__(16) unsigned short Pb[16 * 32];     // 1KB probs bf16
    __shared__ float mrow[16], lrow[16], arow[16];

    const int tid  = threadIdx.x;
    const int wave = tid >> 5;
    const int lane = tid & 31;
    const int lo   = lane & 15;
    const int hi   = lane >> 4;

    const int batch = blockIdx.x >> 7;          // / (2048/16)
    const int qBase = (blockIdx.x & 127) * 16;
    const int qrow0 = batch * S_N + qBase;

    // Stage Q tile asynchronously: 16x512 bf16 = 1024 x b128, 4 per thread.
    #pragma unroll
    for (int j = 0; j < 4; ++j) {
        int idx = tid * 4 + j;
        int r   = idx >> 6;
        int g   = idx & 63;
        unsigned long long gaddr =
            (unsigned long long)(const void*)(Qws + (size_t)(qrow0 + r) * D_N + g * 8);
        unsigned lds_off = (unsigned)(unsigned long long)(const void*)&Qtile[r * D_N + g * 8];
        asm volatile("global_load_async_to_lds_b128 %0, %1, off"
                     :: "v"(lds_off), "v"(gaddr) : "memory");
    }
    if (tid < 16) { mrow[tid] = -3.0e38f; lrow[tid] = 0.f; }
    asm volatile("s_wait_asynccnt 0x0" ::: "memory");  // this wave's async copies done

    const v8f zf = {0.f,0.f,0.f,0.f,0.f,0.f,0.f,0.f};
    v8f accO[4] = {zf, zf, zf, zf};
    const float scale = 0.044194173824159216f;  // 1/sqrt(512)

    for (int kb = 0; kb < S_N; kb += 32) {
        const int krow0 = batch * S_N + kb;
        __syncthreads();   // B0: Qtile ready (iter 0); prev Sfull/Pb consumed

        // Zero score tile
        Sfull[tid] = 0.f;
        Sfull[tid + 256] = 0.f;

        // Prefetch next key block's K / V^T streams (SE scope -> GL2)
        if (kb + 32 < S_N) {
            __builtin_prefetch(Kws + (size_t)(krow0 + 32 + lane) * D_N + wave * 64, 0, 3);
            __builtin_prefetch(Vtws + ((size_t)batch * D_N + wave * 64 + lane) * S_N + kb + 32, 0, 3);
        }

        // Partial scores over this wave's 64-wide D slice: S += Q_slice * K_slice^T
        v8f sacc[2] = {zf, zf};
        #pragma unroll
        for (int ks = 0; ks < 2; ++ks) {
            int d0 = wave * 64 + ks * 32;
            FragBF qa;
            const unsigned short* qp = &Qtile[lo * D_N + d0];
            qa.q[0] = *(const uint4*)(qp + hi * 16);
            qa.q[1] = *(const uint4*)(qp + hi * 16 + 8);
            #pragma unroll
            for (int nt = 0; nt < 2; ++nt) {
                FragBF kf;   // B element (dk,n) = K[kb+n][d0+dk]; contiguous in dk
                const unsigned short* kp =
                    Kws + (size_t)(krow0 + nt * 16 + lo) * D_N + d0 + hi * 16;
                kf.q[0] = *(const uint4*)(kp);
                kf.q[1] = *(const uint4*)(kp + 8);
                sacc[nt] = __builtin_amdgcn_wmma_f32_16x16x32_bf16(
                    false, qa.v, false, kf.v, (short)0, sacc[nt], false, false);
            }
        }
        __syncthreads();   // B1: Sfull zeroed

        // Cross-wave reduction of partial scores via LDS float atomics
        #pragma unroll
        for (int nt = 0; nt < 2; ++nt)
            #pragma unroll
            for (int r = 0; r < 8; ++r)
                atomicAdd(&Sfull[(r + 8 * hi) * 32 + nt * 16 + lo], sacc[nt][r]);
        __syncthreads();   // B2: Sfull complete

        // Online softmax update (one lane per query row)
        if (tid < 16) {
            float mOld = mrow[tid];
            float mx = mOld;
            #pragma unroll
            for (int i = 0; i < 32; ++i)
                mx = fmaxf(mx, Sfull[tid * 32 + i] * scale);
            float alpha = __expf(mOld - mx);
            float sum = 0.f;
            #pragma unroll
            for (int i = 0; i < 32; ++i) {
                float p = __expf(Sfull[tid * 32 + i] * scale - mx);
                sum += p;
                Pb[tid * 32 + i] = f2bf(p);
            }
            lrow[tid] = lrow[tid] * alpha + sum;
            mrow[tid] = mx;
            arow[tid] = alpha;
        }
        __syncthreads();   // B3: Pb/arow visible

        // Rescale accumulators, then accumulate P * V_block into output slice
        #pragma unroll
        for (int nt = 0; nt < 4; ++nt)
            #pragma unroll
            for (int r = 0; r < 8; ++r)
                accO[nt][r] *= arow[r + 8 * hi];

        FragBF pf;
        const unsigned short* pp = &Pb[lo * 32];
        pf.q[0] = *(const uint4*)(pp + hi * 16);
        pf.q[1] = *(const uint4*)(pp + hi * 16 + 8);
        #pragma unroll
        for (int nt = 0; nt < 4; ++nt) {
            FragBF vb;   // B element (kd,n) = V[kb+kd][n] = Vt[b][n][kb+kd]; contiguous in kd
            const unsigned short* vp =
                Vtws + ((size_t)batch * D_N + wave * 64 + nt * 16 + lo) * S_N + kb + hi * 16;
            vb.q[0] = *(const uint4*)(vp);
            vb.q[1] = *(const uint4*)(vp + 8);
            accO[nt] = __builtin_amdgcn_wmma_f32_16x16x32_bf16(
                false, pf.v, false, vb.v, (short)0, accO[nt], false, false);
        }
    }
    __syncthreads();

    // Epilogue: out = (acc / l) * mask
    #pragma unroll
    for (int r = 0; r < 8; ++r) {
        int qr = qBase + r + 8 * hi;
        float inv = maskp[(size_t)batch * S_N + qr] / lrow[r + 8 * hi];
        #pragma unroll
        for (int nt = 0; nt < 4; ++nt) {
            Out[((size_t)batch * S_N + qr) * D_N + wave * 64 + nt * 16 + lo] =
                accO[nt][r] * inv;
        }
    }
}

// ---------------------------------------------------------------------------
extern "C" void kernel_launch(void* const* d_in, const int* in_sizes, int n_in,
                              void* d_out, int out_size, void* d_ws, size_t ws_size,
                              hipStream_t stream) {
    (void)in_sizes; (void)n_in; (void)out_size; (void)ws_size;
    const float* x    = (const float*)d_in[0];
    const float* mask = (const float*)d_in[1];
    const float* Wq   = (const float*)d_in[2];
    const float* bq   = (const float*)d_in[3];
    const float* Wk   = (const float*)d_in[4];
    const float* bk   = (const float*)d_in[5];
    const float* Wv   = (const float*)d_in[6];
    const float* bv   = (const float*)d_in[7];
    float* out = (float*)d_out;

    unsigned short* Qws  = (unsigned short*)d_ws;                 // bf16 [16384,512]
    unsigned short* Kws  = Qws + (size_t)ROWS_N * D_N;
    unsigned short* Vtws = Kws + (size_t)ROWS_N * D_N;            // bf16 [8][512][2048]

    dim3 pgrid(D_N / 64, ROWS_N / 128, 1);   // 8 x 128
    qkv_proj_kernel<false><<<pgrid, 256, 0, stream>>>(x, Wq, bq, Qws);
    qkv_proj_kernel<false><<<pgrid, 256, 0, stream>>>(x, Wk, bk, Kws);
    qkv_proj_kernel<true ><<<pgrid, 256, 0, stream>>>(x, Wv, bv, Vtws);

    attn_kernel<<<dim3(ROWS_N / 16), 256, 0, stream>>>(Qws, Kws, Vtws, mask, out);
}
